// Attention_80908593922315
// MI455X (gfx1250) — matrix operand order repaired and checked by reading, not compiled, not run on hardware
//
#include <hip/hip_runtime.h>
#include <hip/hip_bf16.h>

typedef __attribute__((ext_vector_type(16))) _Float16 v16h;
typedef __attribute__((ext_vector_type(8)))  float    v8f;
typedef __attribute__((ext_vector_type(4)))  int      v4i;

#define C_DIM    256
#define N_SPAT   4096
#define NH       8
#define HD       32
#define THREE_C  768
#define ATTN_SCALE 0.17677669529663687f   // 1/sqrt(32)

// ---------------- workspace layout (bytes) ----------------
#define WS_Y16T  (size_t)0                      // [N][C] 4096*256 f16 = 2 MB
#define WS_WQ16  (size_t)2097152                // 768*256  f16
#define WS_WP16  (size_t)2490368                // 256*256  f16
#define WS_Q16   (size_t)2621440                // [h][n][d] f16
#define WS_K16   (size_t)4718592                // [h][n][d] f16
#define WS_VT16  (size_t)6815744                // [h][d][n] f16 (transposed V)
#define WS_O16T  (size_t)8912896                // attn out as [N][C] f16

// ============ gfx1250 async global->LDS copy (guarded, with fallback) ======
#if defined(__AMDGCN__) && __has_builtin(__builtin_amdgcn_global_load_async_to_lds_b128)
#define HAS_ASYNC_LDS 1
typedef __attribute__((address_space(1))) v4i gas_v4i;
typedef __attribute__((address_space(3))) v4i las_v4i;
__device__ __forceinline__ void async_copy16(const _Float16* g, _Float16* l) {
  __builtin_amdgcn_global_load_async_to_lds_b128((gas_v4i*)(void*)g,
                                                 (las_v4i*)(void*)l, 0, 0);
}
__device__ __forceinline__ void async_wait0() {
#if __has_builtin(__builtin_amdgcn_s_wait_asynccnt)
  __builtin_amdgcn_s_wait_asynccnt(0);
#else
  asm volatile("s_wait_asynccnt 0x0" ::: "memory");
#endif
}
#else
#define HAS_ASYNC_LDS 0
#endif

// ================= WMMA fragment packing (ISA 7.12.2) =================
// A 16x32 f16 from row-major [M][K]: two contiguous 16B chunks -> 2x ds_load_b128
__device__ __forceinline__ v16h pack_A(const _Float16* src, int stride, int lane) {
  const int r   = lane & 15;
  const int klo = (lane < 16) ? 0 : 8;
  v16h a;
#pragma unroll
  for (int e = 0; e < 16; ++e) {
    const int k = (e < 8) ? (klo + e) : (klo + 8 + e);
    a[e] = src[r * stride + k];
  }
  return a;
}
// B 32x16 f16 (K x N) from row-major [N][K] memory: two contiguous 16B chunks/lane
__device__ __forceinline__ v16h pack_BT(const _Float16* src, int stride, int lane) {
  const int n    = lane & 15;
  const int koff = (lane < 16) ? 0 : 8;
  v16h b;
#pragma unroll
  for (int e = 0; e < 16; ++e)
    b[e] = src[n * stride + ((e < 8) ? (koff + e) : (koff + 8 + e))];
  return b;
}

// ================= Kernel 1: weight f32 -> f16 =================
__global__ void cvt_weights_kernel(const float* __restrict__ wq,
                                   const float* __restrict__ wp,
                                   _Float16* __restrict__ wq16,
                                   _Float16* __restrict__ wp16) {
  const int i = blockIdx.x * blockDim.x + threadIdx.x;
  const int NQ = THREE_C * C_DIM;
  if (i < NQ)            wq16[i]      = (_Float16)wq[i];
  else                   wp16[i - NQ] = (_Float16)wp[i - NQ];
}

// ================= Kernel 2: depthwise conv + instance norm ============
__global__ void dwconv_inorm_kernel(const float* __restrict__ x,
                                    const float* __restrict__ w_dw,
                                    const float* __restrict__ b_dw,
                                    _Float16* __restrict__ y16T) {
  const int c   = blockIdx.x;
  const int tid = threadIdx.x;
  __shared__ float xs[N_SPAT];
  __shared__ float wf[27];
  __shared__ float red[16];
  __shared__ float stats[2];

  const float* xc = x + (size_t)c * N_SPAT;
#pragma unroll
  for (int j = 0; j < 16; ++j) xs[tid + 256 * j] = xc[tid + 256 * j];
  if (tid < 27) wf[tid] = w_dw[c * 27 + tid];
  __syncthreads();

  const float bias = b_dw[c];
  float r[16];
  float s = 0.f, ss = 0.f;
#pragma unroll
  for (int j = 0; j < 16; ++j) {
    const int n = tid + 256 * j;
    const int h = n >> 8, w = (n >> 4) & 15, d = n & 15;
    float acc = bias;
#pragma unroll
    for (int kh = 0; kh < 3; ++kh)
#pragma unroll
      for (int kw = 0; kw < 3; ++kw)
#pragma unroll
        for (int kd = 0; kd < 3; ++kd) {
          const int hh = h + kh - 1, ww = w + kw - 1, dd = d + kd - 1;
          if ((unsigned)hh < 16u && (unsigned)ww < 16u && (unsigned)dd < 16u)
            acc += wf[(kh * 3 + kw) * 3 + kd] * xs[(hh << 8) + (ww << 4) + dd];
        }
    r[j] = acc; s += acc; ss += acc * acc;
  }
#pragma unroll
  for (int m = 1; m < 32; m <<= 1) {
    s  += __shfl_xor(s,  m, 32);
    ss += __shfl_xor(ss, m, 32);
  }
  if ((tid & 31) == 0) { red[tid >> 5] = s; red[8 + (tid >> 5)] = ss; }
  __syncthreads();
  if (tid == 0) {
    float S = 0.f, SS = 0.f;
    for (int i = 0; i < 8; ++i) { S += red[i]; SS += red[8 + i]; }
    const float mean = S * (1.f / N_SPAT);
    const float var  = SS * (1.f / N_SPAT) - mean * mean;
    stats[0] = mean;
    stats[1] = rsqrtf(var + 1e-5f);
  }
  __syncthreads();
  const float mean = stats[0], rstd = stats[1];
#pragma unroll
  for (int j = 0; j < 16; ++j) {
    const int n = tid + 256 * j;
    y16T[(size_t)n * C_DIM + c] = (_Float16)((r[j] - mean) * rstd);
  }
}

// ============ shared GEMM core: 64x64 tile, K=256, double-buffered ========
template <typename EPILOG>
__device__ __forceinline__ void gemm_64x64(const _Float16* __restrict__ gAbase,
                                           const _Float16* __restrict__ gBbase,
                                           int tid, EPILOG epilog) {
  const int lane = tid & 31;
  const int wv   = tid >> 5;
  const int wm   = wv >> 1;
  const int wn   = wv & 1;
  __shared__ __align__(16) _Float16 As[2][64 * 32];
  __shared__ __align__(16) _Float16 Bs[2][64 * 32];

  const int hi = tid * 8, row = hi >> 5, col = hi & 31;
  const _Float16* gA = gAbase + (size_t)row * C_DIM + col;
  const _Float16* gB = gBbase + (size_t)row * C_DIM + col;

  // preload tile 0
#if HAS_ASYNC_LDS
  async_copy16(gA, &As[0][hi]);
  async_copy16(gB, &Bs[0][hi]);
  async_wait0();
#else
  *(float4*)&As[0][hi] = *(const float4*)gA;
  *(float4*)&Bs[0][hi] = *(const float4*)gB;
#endif
  __syncthreads();

  v8f acc0 = {}, acc1 = {};
  int cur = 0;
  for (int k0 = 0; k0 < C_DIM; k0 += 32) {
    const int  nxt  = cur ^ 1;
    const bool more = (k0 + 32) < C_DIM;
    if (k0 + 64 < C_DIM) {                       // warm L2 two tiles ahead
      __builtin_prefetch(gA + k0 + 64, 0, 0);
      __builtin_prefetch(gB + k0 + 64, 0, 0);
    }
#if HAS_ASYNC_LDS
    if (more) {                                  // fill next buffer, overlapped
      async_copy16(gA + k0 + 32, &As[nxt][hi]);
      async_copy16(gB + k0 + 32, &Bs[nxt][hi]);
    }
#endif
    const v16h a  = pack_A(&As[cur][wm * 16 * 32], 32, lane);
    const v16h b0 = pack_BT(&Bs[cur][(wn * 32 + 0) * 32],  32, lane);
    const v16h b1 = pack_BT(&Bs[cur][(wn * 32 + 16) * 32], 32, lane);
    acc0 = __builtin_amdgcn_wmma_f32_16x16x32_f16(false, a, false, b0,
                                                  (short)0, acc0, false, false);
    acc1 = __builtin_amdgcn_wmma_f32_16x16x32_f16(false, a, false, b1,
                                                  (short)0, acc1, false, false);
#if HAS_ASYNC_LDS
    async_wait0();
#else
    if (more) {
      *(float4*)&As[nxt][hi] = *(const float4*)(gA + k0 + 32);
      *(float4*)&Bs[nxt][hi] = *(const float4*)(gB + k0 + 32);
    }
#endif
    __syncthreads();
    cur = nxt;
  }
  const int hlf = lane >> 4;
  epilog(acc0, acc1, wm, wn, lane, hlf);
}

// ================= Kernel 3: qkv GEMM (768x4096, K=256) ===============
__global__ void qkv_gemm_kernel(const _Float16* __restrict__ wq16,
                                const _Float16* __restrict__ y16T,
                                const float* __restrict__ b_qkv,
                                _Float16* __restrict__ q16,
                                _Float16* __restrict__ k16,
                                _Float16* __restrict__ vT16) {
  const int o0 = blockIdx.x * 64;
  const int n0 = blockIdx.y * 64;
  gemm_64x64(wq16 + (size_t)o0 * C_DIM, y16T + (size_t)n0 * C_DIM, threadIdx.x,
    [&](const v8f& acc0, const v8f& acc1, int wm, int wn, int lane, int hlf) {
#pragma unroll
      for (int j = 0; j < 2; ++j) {
        const v8f acc = j ? acc1 : acc0;
        const int ncol = n0 + wn * 32 + j * 16 + (lane & 15);
#pragma unroll
        for (int g = 0; g < 8; ++g) {
          const int o = o0 + wm * 16 + g + 8 * hlf;
          float v = acc[g] + b_qkv[o];
          const int s = o >> 8, rem = o & 255;
          const int h = rem >> 5, d = rem & 31;
          if (s == 0) {
            v *= ATTN_SCALE;
            q16[((size_t)h * N_SPAT + ncol) * HD + d] = (_Float16)v;
          } else if (s == 1) {
            k16[((size_t)h * N_SPAT + ncol) * HD + d] = (_Float16)v;
          } else {
            vT16[((size_t)h * HD + d) * N_SPAT + ncol] = (_Float16)v;  // [h][d][n]
          }
        }
      }
    });
}

// ================= Kernel 5: output projection (256x4096, K=256) =======
__global__ void proj_gemm_kernel(const _Float16* __restrict__ wp16,
                                 const _Float16* __restrict__ o16T,
                                 const float* __restrict__ b_proj,
                                 float* __restrict__ out) {
  const int c0 = blockIdx.x * 64;
  const int n0 = blockIdx.y * 64;
  gemm_64x64(wp16 + (size_t)c0 * C_DIM, o16T + (size_t)n0 * C_DIM, threadIdx.x,
    [&](const v8f& acc0, const v8f& acc1, int wm, int wn, int lane, int hlf) {
#pragma unroll
      for (int j = 0; j < 2; ++j) {
        const v8f acc = j ? acc1 : acc0;
        const int ncol = n0 + wn * 32 + j * 16 + (lane & 15);
#pragma unroll
        for (int g = 0; g < 8; ++g) {
          const int c = c0 + wm * 16 + g + 8 * hlf;
          out[(size_t)c * N_SPAT + ncol] = acc[g] + b_proj[c];
        }
      }
    });
}

// ================= Kernel 4: flash attention (O^T = V^T x P^T) =========
__global__ void flash_attn_kernel(const _Float16* __restrict__ q16,
                                  const _Float16* __restrict__ k16,
                                  const _Float16* __restrict__ vT16,
                                  _Float16* __restrict__ o16T) {
  const int tid  = threadIdx.x;
  const int lane = tid & 31;
  const int wv   = tid >> 5;
  const int h    = blockIdx.x >> 5;
  const int q0   = (blockIdx.x & 31) * 128 + wv * 16;
  __shared__ __align__(16) _Float16 Kt[2][32 * 32];     // [key][d]
  __shared__ __align__(16) _Float16 Vt[2][32 * 32];     // [d][key]
  __shared__ __align__(16) _Float16 Pst[8][16 * 32];    // per wave: [query][key]
  __shared__ float wrow[8][16];                         // per wave row->col bcast

  // Q tile -> A-layout registers (resident)
  v16h qa;
  {
    const int r   = lane & 15;
    const int klo = (lane < 16) ? 0 : 8;
    const _Float16* qrow = q16 + ((size_t)h * N_SPAT + q0 + r) * HD;
#pragma unroll
    for (int e = 0; e < 16; ++e) {
      const int kk = (e < 8) ? (klo + e) : (klo + 8 + e);
      qa[e] = qrow[kk];
    }
  }
  float mst[8], lst[8];
#pragma unroll
  for (int g = 0; g < 8; ++g) { mst[g] = -1e30f; lst[g] = 0.f; }
  v8f ot0 = {}, ot1 = {};                // O^T accs: rows=d, cols=query
  const int hlf = lane >> 4;

  // cooperative staging: half the block copies K, half copies V^T
  const int t   = tid & 127;
  const int hi  = t * 8;
  const int srow = hi >> 5, scol = hi & 31;
  const bool isK = (tid < 128);
  const _Float16* gsrc = isK
      ? &k16[((size_t)h * N_SPAT + srow) * HD + scol]
      : &vT16[((size_t)h * HD + srow) * N_SPAT + scol];
  const size_t gmul = isK ? (size_t)HD : (size_t)1;     // halfs per key step

  // preload tile 0
#if HAS_ASYNC_LDS
  async_copy16(gsrc, isK ? &Kt[0][hi] : &Vt[0][hi]);
  async_wait0();
#else
  *(float4*)(isK ? &Kt[0][hi] : &Vt[0][hi]) = *(const float4*)gsrc;
#endif
  __syncthreads();

  int cur = 0;
  for (int key0 = 0; key0 < N_SPAT; key0 += 32) {
    const int  nxt  = cur ^ 1;
    const bool more = (key0 + 32) < N_SPAT;
    if (key0 + 64 < N_SPAT)
      __builtin_prefetch(gsrc + (size_t)(key0 + 64) * gmul, 0, 0);
#if HAS_ASYNC_LDS
    if (more)
      async_copy16(gsrc + (size_t)(key0 + 32) * gmul,
                   isK ? &Kt[nxt][hi] : &Vt[nxt][hi]);
#endif

    // S = Q (16x32) x K^T -> two 16x16 f32 accs (rows=query, cols=key)
    const v16h kb0 = pack_BT(&Kt[cur][0],       32, lane);
    const v16h kb1 = pack_BT(&Kt[cur][16 * 32], 32, lane);
    v8f z = {};
    v8f s0 = __builtin_amdgcn_wmma_f32_16x16x32_f16(false, qa, false, kb0,
                                                    (short)0, z, false, false);
    v8f s1 = __builtin_amdgcn_wmma_f32_16x16x32_f16(false, qa, false, kb1,
                                                    (short)0, z, false, false);

    // online softmax per query row (rows live in 16-lane halves)
    _Float16* P = Pst[wv];
#pragma unroll
    for (int g = 0; g < 8; ++g) {
      float mx = fmaxf(s0[g], s1[g]);
      mx = fmaxf(mx, __shfl_xor(mx, 1, 32));
      mx = fmaxf(mx, __shfl_xor(mx, 2, 32));
      mx = fmaxf(mx, __shfl_xor(mx, 4, 32));
      mx = fmaxf(mx, __shfl_xor(mx, 8, 32));
      const float mnew = fmaxf(mst[g], mx);
      const float corr = __expf(mst[g] - mnew);
      mst[g] = mnew;
      const float p0 = __expf(s0[g] - mnew);
      const float p1 = __expf(s1[g] - mnew);
      float rs = p0 + p1;
      rs += __shfl_xor(rs, 1, 32);
      rs += __shfl_xor(rs, 2, 32);
      rs += __shfl_xor(rs, 4, 32);
      rs += __shfl_xor(rs, 8, 32);
      lst[g] = lst[g] * corr + rs;
      const int prow = g + 8 * hlf;
      if ((lane & 15) == 0) wrow[wv][prow] = corr;     // row -> column broadcast
      P[prow * 32 + (lane & 15)]      = (_Float16)p0;  // P[query][key]
      P[prow * 32 + 16 + (lane & 15)] = (_Float16)p1;
    }
    // per-wave DS ops are in order: safe to read back without a barrier
    const float ccol = wrow[wv][lane & 15];
#pragma unroll
    for (int g = 0; g < 8; ++g) { ot0[g] *= ccol; ot1[g] *= ccol; }

    // O^T += V^T (16d x 32key, pack_A) x P^T (32key x 16query, pack_BT)
    const v16h pb  = pack_BT(P, 32, lane);
    const v16h va0 = pack_A(&Vt[cur][0],       32, lane);
    const v16h va1 = pack_A(&Vt[cur][16 * 32], 32, lane);
    ot0 = __builtin_amdgcn_wmma_f32_16x16x32_f16(false, va0, false, pb,
                                                 (short)0, ot0, false, false);
    ot1 = __builtin_amdgcn_wmma_f32_16x16x32_f16(false, va1, false, pb,
                                                 (short)0, ot1, false, false);

#if HAS_ASYNC_LDS
    async_wait0();
#else
    if (more)
      *(float4*)(isK ? &Kt[nxt][hi] : &Vt[nxt][hi]) =
          *(const float4*)(gsrc + (size_t)(key0 + 32) * gmul);
#endif
    __syncthreads();
    cur = nxt;
  }

  // normalize by per-query l (broadcast row->col through LDS), store [N][C]
#pragma unroll
  for (int g = 0; g < 8; ++g)
    if ((lane & 15) == 0) wrow[wv][g + 8 * hlf] = 1.f / lst[g];
  const float linv = wrow[wv][lane & 15];
  const int n = q0 + (lane & 15);
#pragma unroll
  for (int g = 0; g < 8; ++g) {
    const int c0 = h * HD + g + 8 * hlf;
    const int c1 = c0 + 16;
    o16T[(size_t)n * C_DIM + c0] = (_Float16)(ot0[g] * linv);
    o16T[(size_t)n * C_DIM + c1] = (_Float16)(ot1[g] * linv);
  }
}

// ================= launch =================
extern "C" void kernel_launch(void* const* d_in, const int* in_sizes, int n_in,
                              void* d_out, int out_size, void* d_ws, size_t ws_size,
                              hipStream_t stream) {
  const float* x      = (const float*)d_in[0];
  const float* w_dw   = (const float*)d_in[1];
  const float* b_dw   = (const float*)d_in[2];
  const float* w_qkv  = (const float*)d_in[3];
  const float* b_qkv  = (const float*)d_in[4];
  const float* w_proj = (const float*)d_in[5];
  const float* b_proj = (const float*)d_in[6];
  float* out = (float*)d_out;

  char* ws = (char*)d_ws;
  _Float16* y16T = (_Float16*)(ws + WS_Y16T);
  _Float16* wq16 = (_Float16*)(ws + WS_WQ16);
  _Float16* wp16 = (_Float16*)(ws + WS_WP16);
  _Float16* q16  = (_Float16*)(ws + WS_Q16);
  _Float16* k16  = (_Float16*)(ws + WS_K16);
  _Float16* vT16 = (_Float16*)(ws + WS_VT16);
  _Float16* o16T = (_Float16*)(ws + WS_O16T);

  cvt_weights_kernel<<<(THREE_C * C_DIM + C_DIM * C_DIM) / 256, 256, 0, stream>>>(
      w_qkv, w_proj, wq16, wp16);
  dwconv_inorm_kernel<<<C_DIM, 256, 0, stream>>>(x, w_dw, b_dw, y16T);
  qkv_gemm_kernel<<<dim3(THREE_C / 64, N_SPAT / 64), 256, 0, stream>>>(
      wq16, y16T, b_qkv, q16, k16, vT16);
  flash_attn_kernel<<<NH * (N_SPAT / 128), 256, 0, stream>>>(q16, k16, vT16, o16T);
  proj_gemm_kernel<<<dim3(C_DIM / 64, N_SPAT / 64), 256, 0, stream>>>(
      wp16, o16T, b_proj, out);
}
